// MambaWrap_60284160967100
// MI455X (gfx1250) — matrix operand order repaired and checked by reading, not compiled, not run on hardware
//
#include <hip/hip_runtime.h>
#include <hip/hip_bf16.h>

typedef __attribute__((ext_vector_type(16))) __bf16 v16bf;
typedef __attribute__((ext_vector_type(8)))  float  v8f;

#ifndef __has_builtin
#define __has_builtin(x) 0
#endif
#if __has_builtin(__builtin_amdgcn_global_load_async_to_lds_b128) && \
    __has_builtin(__builtin_amdgcn_s_wait_asynccnt)
#define USE_ASYNC_COPY 1
#else
#define USE_ASYNC_COPY 0
#endif

// ---------------------------------------------------------------- constants
#define BSZ      8
#define LSEQ     2048
#define DMODEL   768
#define DINNER   1536
#define NHEADS   24
#define HDIM     64
#define DSTATE   128
#define CONVDIM  1792
#define NPROJ    3352
#define NPAD     3456      // padded in-proj N (27 * 128)
#define NROWS    (BSZ * LSEQ)   // 16384

// WMMA wrapper: D = A(16x32 bf16) * B(32x16 bf16) + C(16x16 f32)
__device__ __forceinline__ v8f wmma_bf16(v16bf a, v16bf b, v8f c) {
  return __builtin_amdgcn_wmma_f32_16x16x32_bf16(false, a, false, b, (short)0, c,
                                                 false, false);
}

#if USE_ASYNC_COPY
// Builtin signature (from hipcc diagnostic): params are int4 pointers,
// global (addrspace 1, prints as __device__) and LDS (addrspace 3).
typedef int v4i_ __attribute__((vector_size(16)));
typedef __attribute__((address_space(1))) v4i_* g1v4i;
typedef __attribute__((address_space(3))) v4i_* l3v4i;
// 16B per-lane async global->LDS copy, tracked by ASYNCcnt.
// LDS operand is the 32-bit LDS byte offset (low 32 bits of the generic addr
// per the aperture mapping); global operand is the plain 64-bit address.
__device__ __forceinline__ void async_copy16(const void* gsrc, void* ldst) {
  __builtin_amdgcn_global_load_async_to_lds_b128(
      (g1v4i)(unsigned long long)gsrc,
      (l3v4i)(unsigned int)(unsigned long long)ldst, 0, 0);
}
#endif

// A-operand fragment from LDS, A stored row-major [m][k] (k contiguous).
// Lane l holds row (l&15); K block: (l>>4)*16 + e (e linear 0..15).
__device__ __forceinline__ v16bf fragA_ld(const __bf16* lds, int pitch, int row,
                                          int koff, int lane) {
  union { v16bf v; unsigned int u[8]; } f;
  const __bf16* r = lds + row * pitch + koff + ((lane >> 4) << 4);
#pragma unroll
  for (int p = 0; p < 8; ++p)
    f.u[p] = *(const unsigned int*)(r + (p << 1));
  return f.v;
}

// B-operand fragment: B is KxN; LDS stores B^T row-major: [n][k] (k contiguous).
// Lane l holds column (l&15); K block: (l>>4)*16 + e (e linear 0..15).
__device__ __forceinline__ v16bf fragB_ld(const __bf16* lds, int pitch, int col,
                                          int koff, int lane) {
  union { v16bf v; unsigned int u[8]; } f;
  const __bf16* r = lds + col * pitch + koff + ((lane >> 4) << 4);
#pragma unroll
  for (int p = 0; p < 8; ++p)
    f.u[p] = *(const unsigned int*)(r + (p << 1));
  return f.v;
}

// B-operand fragment where LDS stores B as [k][n] (n contiguous) -> strided gather
__device__ __forceinline__ v16bf fragB_strided(const __bf16* lds, int pitch, int col,
                                               int koff, int lane) {
  union { v16bf v; __bf16 b[16]; } f;
  int k0 = koff + ((lane >> 4) << 4);
#pragma unroll
  for (int e = 0; e < 16; ++e) f.b[e] = lds[(k0 + e) * pitch + col];
  return f.v;
}

// B-operand fragment from f32 LDS stored as B^T [col][k], converted to bf16
__device__ __forceinline__ v16bf fragB_f32(const float* lds, int pitch, int col,
                                           int koff, int lane) {
  union { v16bf v; __bf16 b[16]; } f;
  const float* r = lds + col * pitch + koff + ((lane >> 4) << 4);
#pragma unroll
  for (int e = 0; e < 16; ++e) f.b[e] = (__bf16)r[e];
  return f.v;
}

// ---------------------------------------------------------------- LayerNorm
__global__ __launch_bounds__(256) void ln_kernel(const float* __restrict__ x,
                                                 const float* __restrict__ w,
                                                 const float* __restrict__ bb,
                                                 __bf16* __restrict__ xn) {
  __shared__ float red[256];
  const int row = blockIdx.x;
  const float* xr = x + (size_t)row * DMODEL;
  float s = 0.f;
  for (int i = threadIdx.x; i < DMODEL; i += 256) s += xr[i];
  red[threadIdx.x] = s; __syncthreads();
  for (int o = 128; o > 0; o >>= 1) {
    if (threadIdx.x < o) red[threadIdx.x] += red[threadIdx.x + o];
    __syncthreads();
  }
  float mu = red[0] / DMODEL; __syncthreads();
  float v = 0.f;
  for (int i = threadIdx.x; i < DMODEL; i += 256) { float d = xr[i] - mu; v += d * d; }
  red[threadIdx.x] = v; __syncthreads();
  for (int o = 128; o > 0; o >>= 1) {
    if (threadIdx.x < o) red[threadIdx.x] += red[threadIdx.x + o];
    __syncthreads();
  }
  float rstd = rsqrtf(red[0] / DMODEL + 1e-5f);
  for (int i = threadIdx.x; i < DMODEL; i += 256)
    xn[(size_t)row * DMODEL + i] = (__bf16)((xr[i] - mu) * rstd * w[i] + bb[i]);
}

// -------------------------------------------------- weight transpose/convert
__global__ __launch_bounds__(256) void cvt_win_kernel(const float* __restrict__ W,
                                                      __bf16* __restrict__ WT) {
  int idx = blockIdx.x * 256 + threadIdx.x;
  if (idx >= NPAD * DMODEL) return;
  int n = idx / DMODEL, k = idx % DMODEL;
  WT[idx] = (n < NPROJ) ? (__bf16)W[(size_t)k * NPROJ + n] : (__bf16)0.f;
}
__global__ __launch_bounds__(256) void cvt_wout_kernel(const float* __restrict__ W,
                                                       __bf16* __restrict__ WT) {
  int idx = blockIdx.x * 256 + threadIdx.x;
  if (idx >= DMODEL * DINNER) return;
  int n = idx / DINNER, k = idx % DINNER;
  WT[idx] = (__bf16)W[(size_t)k * DMODEL + n];
}

// ---------------------------------------------------------------- bf16 GEMM
// C[M,N](f32) = A[M,K](bf16) * B[K,N] (stored as BT[N,K] bf16) (+ residual)
// Double-buffered LDS pipeline: tile kk+1 streams in (async or via regs)
// while the 8-WMMA chain for tile kk executes.
#define GTM 128
#define GTK 32
#define APITCH 40   // 20-dword row stride -> conflict-free frags

__global__ __launch_bounds__(256) void gemm_bf16_kernel(
    const __bf16* __restrict__ A, const __bf16* __restrict__ BT,
    const float* __restrict__ residual, float* __restrict__ C,
    int M, int N, int K) {
  __shared__ __bf16 As[2][GTM * APITCH];
  __shared__ __bf16 Bs[2][GTM * APITCH];
  const int tid = threadIdx.x, lane = tid & 31, w = tid >> 5;
  const int bm = blockIdx.x * GTM, bn = blockIdx.y * GTM;

  // staging coordinates: each thread moves 2x16B of A and 2x16B of B per tile
  const int r0 = tid >> 2, s0 = (tid & 3) * 8;
  const int r1 = r0 + 64,  s1 = s0;
  const __bf16* gA0 = A + (size_t)(bm + r0) * K + s0;
  const __bf16* gA1 = A + (size_t)(bm + r1) * K + s1;
  const __bf16* gB0 = BT + (size_t)(bn + r0) * K + s0;
  const __bf16* gB1 = BT + (size_t)(bn + r1) * K + s1;
  __bf16* lA0[2] = { &As[0][r0 * APITCH + s0], &As[1][r0 * APITCH + s0] };
  __bf16* lA1[2] = { &As[0][r1 * APITCH + s1], &As[1][r1 * APITCH + s1] };
  __bf16* lB0[2] = { &Bs[0][r0 * APITCH + s0], &Bs[1][r0 * APITCH + s0] };
  __bf16* lB1[2] = { &Bs[0][r1 * APITCH + s1], &Bs[1][r1 * APITCH + s1] };

  v8f acc[8] = {};
  int cur = 0;

#if USE_ASYNC_COPY
  // prologue: async-stage tile 0 into buffer 0
  async_copy16(gA0, lA0[0]); async_copy16(gA1, lA1[0]);
  async_copy16(gB0, lB0[0]); async_copy16(gB1, lB1[0]);
  for (int kk = 0; kk < K; kk += GTK) {
    __builtin_amdgcn_s_wait_asynccnt(0);   // tile kk landed in LDS
    __syncthreads();
    if (kk + GTK < K) {                    // stream tile kk+1 into other buffer
      int nxt = cur ^ 1;
      async_copy16(gA0 + kk + GTK, lA0[nxt]);
      async_copy16(gA1 + kk + GTK, lA1[nxt]);
      async_copy16(gB0 + kk + GTK, lB0[nxt]);
      async_copy16(gB1 + kk + GTK, lB1[nxt]);
    }
    const __bf16* Ab = As[cur];
    const __bf16* Bb = Bs[cur];
    v16bf af = fragA_ld(Ab, APITCH, w * 16 + (lane & 15), 0, lane);
    v16bf bfr[8];
#pragma unroll
    for (int j = 0; j < 8; ++j)
      bfr[j] = fragB_ld(Bb, APITCH, j * 16 + (lane & 15), 0, lane);
#pragma unroll
    for (int j = 0; j < 8; ++j) acc[j] = wmma_bf16(af, bfr[j], acc[j]);
    cur ^= 1;
  }
#else
  // prologue: load tile 0 into registers
  uint4 ra0 = *(const uint4*)gA0, ra1 = *(const uint4*)gA1;
  uint4 rb0 = *(const uint4*)gB0, rb1 = *(const uint4*)gB1;
  for (int kk = 0; kk < K; kk += GTK) {
    *(uint4*)lA0[cur] = ra0; *(uint4*)lA1[cur] = ra1;
    *(uint4*)lB0[cur] = rb0; *(uint4*)lB1[cur] = rb1;
    __syncthreads();
    if (kk + GTK < K) {   // next tile's global loads overlap the WMMA chain
      ra0 = *(const uint4*)(gA0 + kk + GTK);
      ra1 = *(const uint4*)(gA1 + kk + GTK);
      rb0 = *(const uint4*)(gB0 + kk + GTK);
      rb1 = *(const uint4*)(gB1 + kk + GTK);
      if (kk + 2 * GTK < K) {
        __builtin_prefetch(gA0 + kk + 2 * GTK, 0, 0);
        __builtin_prefetch(gB0 + kk + 2 * GTK, 0, 0);
      }
    }
    const __bf16* Ab = As[cur];
    const __bf16* Bb = Bs[cur];
    v16bf af = fragA_ld(Ab, APITCH, w * 16 + (lane & 15), 0, lane);
    v16bf bfr[8];
#pragma unroll
    for (int j = 0; j < 8; ++j)
      bfr[j] = fragB_ld(Bb, APITCH, j * 16 + (lane & 15), 0, lane);
#pragma unroll
    for (int j = 0; j < 8; ++j) acc[j] = wmma_bf16(af, bfr[j], acc[j]);
    cur ^= 1;
  }
#endif

  const int mrow = bm + w * 16;
#pragma unroll
  for (int j = 0; j < 8; ++j) {
    int n = bn + j * 16 + (lane & 15);
#pragma unroll
    for (int r = 0; r < 8; ++r) {
      int m = mrow + r + ((lane >> 4) << 3);
      float v = acc[j][r];
      if (residual) v += residual[(size_t)m * N + n];
      C[(size_t)m * N + n] = v;
    }
  }
}

// ------------------------------------------------------- conv1d(w=4) + SiLU
__global__ __launch_bounds__(256) void conv_kernel(const float* __restrict__ zx,
                                                   const float* __restrict__ cw,
                                                   const float* __restrict__ cb,
                                                   __bf16* __restrict__ xsOut,
                                                   __bf16* __restrict__ Bc,
                                                   __bf16* __restrict__ Cc) {
  const int bl = blockIdx.x;
  const int b = bl / LSEQ, l = bl % LSEQ;
  for (int ci = 0; ci < 7; ++ci) {
    int c = ci * 256 + threadIdx.x;     // CONVDIM = 7*256
    float acc = cb[c];
#pragma unroll
    for (int k = 0; k < 4; ++k) {
      int lk = l - 3 + k;
      if (lk >= 0)
        acc += zx[((size_t)b * LSEQ + lk) * NPAD + DINNER + c] * cw[c * 4 + k];
    }
    float sv = acc / (1.f + __expf(-acc));
    if (c < DINNER) {
      int h = c >> 6, p = c & 63;
      xsOut[(((size_t)(b * NHEADS + h) * LSEQ) + l) * HDIM + p] = (__bf16)sv;
    } else if (c < DINNER + DSTATE) {
      Bc[((size_t)b * LSEQ + l) * DSTATE + (c - DINNER)] = (__bf16)sv;
    } else {
      Cc[((size_t)b * LSEQ + l) * DSTATE + (c - DINNER - DSTATE)] = (__bf16)sv;
    }
  }
}

// ----------------------------------------------------- dt = softplus(dt + b)
__global__ __launch_bounds__(256) void dt_kernel(const float* __restrict__ zx,
                                                 const float* __restrict__ dtb,
                                                 float* __restrict__ dtf) {
  int idx = blockIdx.x * 256 + threadIdx.x;     // (b,h,l) flat
  if (idx >= BSZ * NHEADS * LSEQ) return;
  int l = idx % LSEQ; int bh = idx / LSEQ; int h = bh % NHEADS; int b = bh / NHEADS;
  float v = zx[((size_t)b * LSEQ + l) * NPAD + (DINNER + CONVDIM) + h] + dtb[h];
  dtf[idx] = (v > 20.f) ? v : log1pf(__expf(v));
}

// ---------------------------------------------------- SSD chunked scan (WMMA)
#define QC     64
#define PIT64  72     // 36-dword stride, gcd(36,64)=4 -> 16 distinct banks
#define PIT128 136    // 68-dword stride
#define PITS   132    // f32 state pitch

__global__ __launch_bounds__(128) void scan_kernel(const __bf16* __restrict__ xs,
                                                   const __bf16* __restrict__ Bc,
                                                   const __bf16* __restrict__ Cc,
                                                   const float* __restrict__ dtg,
                                                   const float* __restrict__ A_log,
                                                   const float* __restrict__ Dv,
                                                   float* __restrict__ yout) {
  extern __shared__ char smem[];
  __bf16* Xs  = (__bf16*)smem;              // [QC][PIT64]   X chunk (t,p)
  __bf16* XwT = Xs  + QC * PIT64;           // [P][PIT64]    (dt*decay*x)^T (p,t)
  __bf16* Mm  = XwT + QC * PIT64;           // [QC][PIT64]   masked decay matrix
  __bf16* Bt  = Mm  + QC * PIT64;           // [QC][PIT128]  B chunk (t,n)
  __bf16* Ct  = Bt  + QC * PIT128;          // [QC][PIT128]  C chunk (t,n)
  float*  Sf  = (float*)(Ct + QC * PIT128); // [P][PITS]     state S (p,n) f32
  float*  cs  = Sf + HDIM * PITS;           // [QC] inclusive cumsum of dt*A
  float*  dtv = cs + QC;                    // [QC]

  const int tid = threadIdx.x, lane = tid & 31, w = tid >> 5;
  const int bh = blockIdx.x;
  const int h = bh % NHEADS, b = bh / NHEADS;
  const float Ah = -__expf(A_log[h]);
  const float Dh = Dv[h];
  const __bf16* gX = xs + (size_t)bh * LSEQ * HDIM;
  const float*  gdt = dtg + (size_t)bh * LSEQ;
  float* gy = yout + (size_t)bh * LSEQ * HDIM;

  for (int i = tid; i < HDIM * PITS; i += 128) Sf[i] = 0.f;
  __syncthreads();

  const int tb = w * 16;
  const int rowA = tb + (lane & 15);

  for (int c = 0; c < LSEQ / QC; ++c) {
    const int l0 = c * QC;
    // --- dt, a = dt*A, inclusive cumsum (Hillis-Steele over 64) ------------
    if (tid < QC) { float d = gdt[l0 + tid]; dtv[tid] = d; cs[tid] = d * Ah; }
    __syncthreads();
    for (int off = 1; off < QC; off <<= 1) {
      float v = 0.f;
      if (tid < QC && tid >= off) v = cs[tid - off];
      __syncthreads();
      if (tid < QC) cs[tid] += v;
      __syncthreads();
    }
    // --- load X (64x64), B/C (64x128) via dword copies ---------------------
    {
      const unsigned int* gx32 = (const unsigned int*)(gX + (size_t)l0 * HDIM);
      for (int i = tid; i < QC * HDIM / 2; i += 128) {
        int t = i >> 5, pp = (i & 31) << 1;
        *(unsigned int*)(Xs + t * PIT64 + pp) = gx32[i];
      }
      const unsigned int* gb32 =
          (const unsigned int*)(Bc + ((size_t)b * LSEQ + l0) * DSTATE);
      const unsigned int* gc32 =
          (const unsigned int*)(Cc + ((size_t)b * LSEQ + l0) * DSTATE);
      for (int i = tid; i < QC * DSTATE / 2; i += 128) {
        int t = i >> 6, nn = (i & 63) << 1;
        *(unsigned int*)(Bt + t * PIT128 + nn) = gb32[i];
        *(unsigned int*)(Ct + t * PIT128 + nn) = gc32[i];
      }
    }
    __syncthreads();
    const float e_end = __expf(cs[QC - 1]);
    // --- XwT[p][t] = x[t][p] * dt[t] * exp(cs_end - cs[t]) -----------------
    for (int i = tid; i < QC * HDIM; i += 128) {
      int t = i >> 6, p = i & 63;
      float xv = (float)Xs[t * PIT64 + p];
      XwT[p * PIT64 + t] = (__bf16)(xv * dtv[t] * __expf(cs[QC - 1] - cs[t]));
    }
    __syncthreads();
    // --- G = C * B^T (64x64, K=128); mask+decay -> Mm ----------------------
    {
      v8f g[4] = {};
#pragma unroll
      for (int ks = 0; ks < DSTATE; ks += 32) {
        v16bf af = fragA_ld(Ct, PIT128, rowA, ks, lane);
        v16bf bfr[4];
#pragma unroll
        for (int j = 0; j < 4; ++j)
          bfr[j] = fragB_ld(Bt, PIT128, j * 16 + (lane & 15), ks, lane);
#pragma unroll
        for (int j = 0; j < 4; ++j) g[j] = wmma_bf16(af, bfr[j], g[j]);
      }
#pragma unroll
      for (int j = 0; j < 4; ++j) {
        int s = j * 16 + (lane & 15);
#pragma unroll
        for (int r = 0; r < 8; ++r) {
          int t = tb + r + ((lane >> 4) << 3);
          float val = (s <= t) ? g[j][r] * __expf(cs[t] - cs[s]) * dtv[s] : 0.f;
          Mm[t * PIT64 + s] = (__bf16)val;
        }
      }
    }
    __syncthreads();
    // --- scale C rows by exp(cs[t]) in place -------------------------------
    for (int i = tid; i < QC * DSTATE / 2; i += 128) {
      int t = i >> 6, nn = (i & 63) << 1;
      float e = __expf(cs[t]);
      __bf16* pC = Ct + t * PIT128 + nn;
      pC[0] = (__bf16)((float)pC[0] * e);
      pC[1] = (__bf16)((float)pC[1] * e);
    }
    __syncthreads();
    // --- Y = Csc * S^T + M * X ; write y + D*x -----------------------------
    {
      v8f y[4] = {};
#pragma unroll
      for (int ks = 0; ks < DSTATE; ks += 32) {
        v16bf af = fragA_ld(Ct, PIT128, rowA, ks, lane);
        v16bf bfr[4];
#pragma unroll
        for (int j = 0; j < 4; ++j)
          bfr[j] = fragB_f32(Sf, PITS, j * 16 + (lane & 15), ks, lane);
#pragma unroll
        for (int j = 0; j < 4; ++j) y[j] = wmma_bf16(af, bfr[j], y[j]);
      }
#pragma unroll
      for (int ks = 0; ks < QC; ks += 32) {
        v16bf af = fragA_ld(Mm, PIT64, rowA, ks, lane);
        v16bf bfr[4];
#pragma unroll
        for (int j = 0; j < 4; ++j)
          bfr[j] = fragB_strided(Xs, PIT64, j * 16 + (lane & 15), ks, lane);
#pragma unroll
        for (int j = 0; j < 4; ++j) y[j] = wmma_bf16(af, bfr[j], y[j]);
      }
#pragma unroll
      for (int j = 0; j < 4; ++j) {
        int p = j * 16 + (lane & 15);
#pragma unroll
        for (int r = 0; r < 8; ++r) {
          int t = tb + r + ((lane >> 4) << 3);
          float xv = (float)Xs[t * PIT64 + p];
          gy[(size_t)(l0 + t) * HDIM + p] = y[j][r] + Dh * xv;
        }
      }
    }
    __syncthreads();
    // --- S = e_end * S + XwT * B (64x128, K=64) ----------------------------
    {
      v8f su[8] = {};
      const int pb = w * 16;
      const int rowP = pb + (lane & 15);
#pragma unroll
      for (int ks = 0; ks < QC; ks += 32) {
        v16bf af = fragA_ld(XwT, PIT64, rowP, ks, lane);
        v16bf bfr[8];
#pragma unroll
        for (int j = 0; j < 8; ++j)
          bfr[j] = fragB_strided(Bt, PIT128, j * 16 + (lane & 15), ks, lane);
#pragma unroll
        for (int j = 0; j < 8; ++j) su[j] = wmma_bf16(af, bfr[j], su[j]);
      }
#pragma unroll
      for (int j = 0; j < 8; ++j) {
        int n = j * 16 + (lane & 15);
#pragma unroll
        for (int r = 0; r < 8; ++r) {
          int p = pb + r + ((lane >> 4) << 3);
          Sf[p * PITS + n] = e_end * Sf[p * PITS + n] + su[j][r];
        }
      }
    }
    __syncthreads();
  }
}

// -------------------------------------------- gating (silu(z)) + RMSNorm
__global__ __launch_bounds__(256) void gate_rms_kernel(const float* __restrict__ yf,
                                                       const float* __restrict__ zx,
                                                       const float* __restrict__ rmsw,
                                                       __bf16* __restrict__ yb) {
  __shared__ float red[256];
  const int row = blockIdx.x;
  const int b = row / LSEQ, l = row % LSEQ;
  float g[6]; float ss = 0.f;
#pragma unroll
  for (int ii = 0; ii < 6; ++ii) {
    int e = ii * 256 + threadIdx.x;
    int h = e >> 6, p = e & 63;
    float yv = yf[(((size_t)(b * NHEADS + h) * LSEQ) + l) * HDIM + p];
    float zv = zx[(size_t)row * NPAD + e];
    float gv = yv * (zv / (1.f + __expf(-zv)));
    g[ii] = gv; ss += gv * gv;
  }
  red[threadIdx.x] = ss; __syncthreads();
  for (int o = 128; o > 0; o >>= 1) {
    if (threadIdx.x < o) red[threadIdx.x] += red[threadIdx.x + o];
    __syncthreads();
  }
  float scale = rsqrtf(red[0] / (float)DINNER + 1e-5f);
#pragma unroll
  for (int ii = 0; ii < 6; ++ii) {
    int e = ii * 256 + threadIdx.x;
    yb[(size_t)row * DINNER + e] = (__bf16)(g[ii] * scale * rmsw[e]);
  }
}

// ---------------------------------------------------------------- launcher
extern "C" void kernel_launch(void* const* d_in, const int* in_sizes, int n_in,
                              void* d_out, int out_size, void* d_ws, size_t ws_size,
                              hipStream_t stream) {
  const float* x      = (const float*)d_in[0];
  const float* norm_w = (const float*)d_in[1];
  const float* norm_b = (const float*)d_in[2];
  const float* W_in   = (const float*)d_in[3];
  const float* conv_w = (const float*)d_in[4];
  const float* conv_b = (const float*)d_in[5];
  const float* dt_b   = (const float*)d_in[6];
  const float* A_log  = (const float*)d_in[7];
  const float* Dvec   = (const float*)d_in[8];
  const float* rms_w  = (const float*)d_in[9];
  const float* W_out  = (const float*)d_in[10];
  float* out = (float*)d_out;

  char* ws = (char*)d_ws;
  size_t off = 0;
  auto take = [&](size_t bytes) { char* p = ws + off; off += (bytes + 255) & ~(size_t)255; return p; };

  __bf16* xn_bf  = (__bf16*)take((size_t)NROWS * DMODEL * 2);
  __bf16* WinT   = (__bf16*)take((size_t)NPAD * DMODEL * 2);
  __bf16* WoutT  = (__bf16*)take((size_t)DMODEL * DINNER * 2);
  float*  zx     = (float*) take((size_t)NROWS * NPAD * 4);
  __bf16* xs_bf  = (__bf16*)take((size_t)BSZ * NHEADS * LSEQ * HDIM * 2);
  __bf16* Bc_bf  = (__bf16*)take((size_t)NROWS * DSTATE * 2);
  __bf16* Cc_bf  = (__bf16*)take((size_t)NROWS * DSTATE * 2);
  float*  dt_f   = (float*) take((size_t)BSZ * NHEADS * LSEQ * 4);
  float*  y_f    = (float*) take((size_t)BSZ * NHEADS * LSEQ * HDIM * 4);
  __bf16* y_bf   = (__bf16*)take((size_t)NROWS * DINNER * 2);
  (void)ws_size; (void)n_in; (void)in_sizes; (void)out_size;

  // 1) LayerNorm -> bf16 activations
  ln_kernel<<<NROWS, 256, 0, stream>>>(x, norm_w, norm_b, xn_bf);
  // 2) weight transposes -> bf16
  cvt_win_kernel<<<(NPAD * DMODEL + 255) / 256, 256, 0, stream>>>(W_in, WinT);
  cvt_wout_kernel<<<(DMODEL * DINNER + 255) / 256, 256, 0, stream>>>(W_out, WoutT);
  // 3) in-proj GEMM: zx[16384, 3456] = xn * W_in
  gemm_bf16_kernel<<<dim3(NROWS / GTM, NPAD / GTM), 256, 0, stream>>>(
      xn_bf, WinT, nullptr, zx, NROWS, NPAD, DMODEL);
  // 4) causal depthwise conv + SiLU, split into xs / B / C
  conv_kernel<<<NROWS, 256, 0, stream>>>(zx, conv_w, conv_b, xs_bf, Bc_bf, Cc_bf);
  // 5) dt softplus
  dt_kernel<<<(BSZ * NHEADS * LSEQ + 255) / 256, 256, 0, stream>>>(zx, dt_b, dt_f);
  // 6) SSD chunked selective scan (WMMA)
  {
    size_t smem = (size_t)(3 * QC * PIT64 + 2 * QC * PIT128) * 2 +
                  (size_t)(HDIM * PITS + 2 * QC) * 4;
    scan_kernel<<<BSZ * NHEADS, 128, smem, stream>>>(xs_bf, Bc_bf, Cc_bf, dt_f,
                                                     A_log, Dvec, y_f);
  }
  // 7) gating + RMSNorm -> bf16
  gate_rms_kernel<<<NROWS, 256, 0, stream>>>(y_f, zx, rms_w, y_bf);
  // 8) out-proj GEMM + residual
  gemm_bf16_kernel<<<dim3(NROWS / GTM, DMODEL / GTM), 256, 0, stream>>>(
      y_bf, WoutT, x, out, NROWS, DMODEL, DINNER);
}
